// NeuralSplineFlow_81423989997885
// MI455X (gfx1250) — matrix operand order repaired and checked by reading, not compiled, not run on hardware
//
#include <hip/hip_runtime.h>
#include <hip/hip_bf16.h>
#include <math.h>

// ---------------- problem constants ----------------
constexpr int   D_    = 64;
constexpr int   H_    = 512;
constexpr int   NL_   = 8;
constexpr int   K_    = 8;
constexpr int   MULT_ = 23;          // 3K-1
constexpr int   P_    = D_ * MULT_;  // 1472
constexpr long long B_ = 65536;
constexpr float TB_   = 3.0f;
constexpr float MINW_ = 0.001f;
constexpr float MINH_ = 0.001f;
constexpr float MIND_ = 0.001f;

// ---------------- WMMA vector types ----------------
typedef __attribute__((ext_vector_type(16))) __bf16 v16bf;
typedef __attribute__((ext_vector_type(8)))  __bf16 v8bf;
typedef __attribute__((ext_vector_type(8)))  float  v8f;

// ---------------- LDS layout (dynamic) ----------------
constexpr int ZIN_STR = 68;    // f32, z after LU (spline input)
constexpr int XIN_STR = 72;    // bf16 copy of z for GEMM1 A operand
constexpr int H_STR   = 520;   // bf16 hidden activations (padded vs 512)
constexpr int P_STR   = 1480;  // f32 spline params (padded vs 1472)
constexpr int SM_ZIN  = 16 * ZIN_STR * 4;            // 4352
constexpr int SM_XIN  = 16 * XIN_STR * 2;            // 2304
constexpr int SM_H    = 16 * H_STR * 2;              // 16640
constexpr int SM_P    = 16 * P_STR * 4;              // 94720
constexpr int SMEM_TOTAL = SM_ZIN + SM_XIN + 2 * SM_H + SM_P + 64; // 134720 B

// ---------------- workspace layout (bytes) ----------------
constexpr size_t OFF_W0M   = 0;                                 // bf16 8*512*64
constexpr size_t OFF_WHM   = OFF_W0M + 524288;                  // bf16 8*2*512*512
constexpr size_t OFF_WOM   = OFF_WHM + 8388608;                 // bf16 8*1472*512
constexpr size_t OFF_AFOLD = OFF_WOM + 12058624;                // f32  8*64*64
constexpr size_t OFF_CONST = OFF_AFOLD + 131072;                // f32  8
constexpr size_t OFF_LOGQ  = OFF_CONST + 256;                   // f32  65536
constexpr size_t OFF_ZA    = OFF_LOGQ + 262144;                 // f32  65536*64
constexpr size_t OFF_ZB    = OFF_ZA + 16777216;

// =====================================================================
// Prep 1: apply MADE degree masks, convert weights to bf16
// =====================================================================
__global__ void nsf_prep_weights(const float* __restrict__ W0,
                                 const float* __restrict__ Wh,
                                 const float* __restrict__ Wo,
                                 __bf16* __restrict__ W0m,
                                 __bf16* __restrict__ Whm,
                                 __bf16* __restrict__ Wom)
{
    const long long stride = (long long)gridDim.x * blockDim.x;
    const long long gid    = (long long)blockIdx.x * blockDim.x + threadIdx.x;

    const long long n0 = (long long)NL_ * H_ * D_;          // 262144
    for (long long i = gid; i < n0; i += stride) {
        int dcol = (int)(i & 63);
        int h    = (int)((i >> 6) % H_);
        int hd   = (h % (D_ - 1)) + 1;
        float v  = (hd >= dcol + 1) ? W0[i] : 0.0f;
        W0m[i] = (__bf16)v;
    }
    const long long n1 = (long long)NL_ * 2 * H_ * H_;      // 4194304
    for (long long i = gid; i < n1; i += stride) {
        int hc = (int)(i & 511);
        int hr = (int)((i >> 9) % H_);
        float v = (((hr % 63) + 1) >= ((hc % 63) + 1)) ? Wh[i] : 0.0f;
        Whm[i] = (__bf16)v;
    }
    const long long n2 = (long long)NL_ * P_ * H_;          // 6029312
    for (long long i = gid; i < n2; i += stride) {
        int hc = (int)(i & 511);
        int o  = (int)((i >> 9) % P_);
        int od = o / MULT_ + 1;
        float v = (od > ((hc % 63) + 1)) ? Wo[i] : 0.0f;
        Wom[i] = (__bf16)v;
    }
}

// =====================================================================
// Prep 2: per layer, fold permutation into A = (L*U)^T, plus log|det U|
//   z' = z @ Afold,  Afold[j][r] = M[r][invperm[j]],  M = (tril(L,-1)+I)@triu(U)
// =====================================================================
__global__ void nsf_prep_lu(const float* __restrict__ lower,
                            const float* __restrict__ upper,
                            const int*   __restrict__ perm,
                            float* __restrict__ Afold,
                            float* __restrict__ consts)
{
    const int i = blockIdx.x;     // layer
    const int t = threadIdx.x;    // 0..63
    __shared__ float M[64][65];
    __shared__ int invp[64];

    const float* L = lower + (size_t)i * 4096;
    const float* U = upper + (size_t)i * 4096;

    for (int c = 0; c < 64; ++c) {
        float acc = (c >= t) ? U[t * 64 + c] : 0.0f;
        for (int k = 0; k < t; ++k)
            if (c >= k) acc += L[t * 64 + k] * U[k * 64 + c];
        M[t][c] = acc;
    }
    invp[perm[i * 64 + t]] = t;
    __syncthreads();

    const int ip = invp[t];
    for (int r = 0; r < 64; ++r)
        Afold[(size_t)i * 4096 + t * 64 + r] = M[r][ip];

    if (t == 0) {
        float s = 0.0f;
        for (int d = 0; d < 64; ++d) s += logf(fabsf(U[d * 64 + d]));
        consts[i] = s;
    }
}

// =====================================================================
// GEMM stage: out[m,n] = act( sum_k hin[m,k]*W[n,k] + bias[n] )
//   W masked bf16 row-major [Ncols][Kdim]; A from LDS; B streamed from L2.
//   WMMA 16x16x32 bf16 -> f32.  8 waves, 4 n-tiles per wave chunk.
// =====================================================================
template<bool RELU>
__device__ __forceinline__ void made_gemm(
    const __bf16* __restrict__ W, const float* __restrict__ bias,
    int Kdim, int ntiles,
    const __bf16* __restrict__ hin, int hin_str,
    __bf16* __restrict__ hout, int hout_str,
    float*  __restrict__ fout, int fout_str)
{
    const int lane = threadIdx.x & 31;
    const int wave = threadIdx.x >> 5;
    const int m    = lane & 15;
    const int half = lane >> 4;
    const int ncol = lane & 15;
    const int ktiles = Kdim >> 5;

    for (int c = wave; c * 4 < ntiles; c += 8) {
        const int nt0 = c * 4;
        v8f acc[4] = {};
        const __bf16* wp[4];
#pragma unroll
        for (int t = 0; t < 4; ++t)
            wp[t] = W + (size_t)((nt0 + t) * 16 + ncol) * Kdim + 16 * half;

        for (int kt = 0; kt < ktiles; ++kt) {
            const int k0 = kt << 5;
            // A operand: lane holds hin[m][k0+16*half .. +16] — two 16B chunks per lane
            v8bf lo = *(const v8bf*)(hin + m * hin_str + k0 + 16 * half);
            v8bf hi = *(const v8bf*)(hin + m * hin_str + k0 + 16 * half + 8);
            v16bf a;
#pragma unroll
            for (int j = 0; j < 8; ++j) { a[j] = lo[j]; a[j + 8] = hi[j]; }
#pragma unroll
            for (int t = 0; t < 4; ++t) {
                // B operand: lane holds W[n][k0+16*half .. +16] (32 contiguous B)
                v16bf b = *(const v16bf*)(wp[t] + k0);
                acc[t] = __builtin_amdgcn_wmma_f32_16x16x32_bf16(
                    false, a, false, b, (short)0, acc[t], false, false);
            }
        }
#pragma unroll
        for (int t = 0; t < 4; ++t) {
            const int n  = (nt0 + t) * 16 + ncol;
            const float bv = bias[n];
#pragma unroll
            for (int v = 0; v < 8; ++v) {
                const float val = acc[t][v] + bv;
                const int mr = v + 8 * half;      // C/D layout: M = v + 8*half
                if constexpr (RELU)
                    hout[mr * hout_str + n] = (__bf16)fmaxf(val, 0.0f);
                else
                    fout[mr * fout_str + n] = val;
            }
        }
    }
}

__device__ __forceinline__ float softplusf_(float x)
{
    return (x > 20.0f) ? x : log1pf(expf(x));
}

// =====================================================================
// Fused flow layer: perm+LU, MADE (3 GEMMs + output GEMM), RQ spline.
// One block = 16 batch rows, 256 threads (8 wave32).
// =====================================================================
__global__ void __launch_bounds__(256)
nsf_layer(const float* __restrict__ zsrc, float* __restrict__ zdst,
          const __bf16* __restrict__ W0m,
          const __bf16* __restrict__ Wh0, const __bf16* __restrict__ Wh1,
          const __bf16* __restrict__ Wom,
          const float* __restrict__ b0,
          const float* __restrict__ bh0, const float* __restrict__ bh1,
          const float* __restrict__ bo,
          const float* __restrict__ Afold, const float* __restrict__ lubias,
          float* __restrict__ logq, int first)
{
    extern __shared__ char smem_raw[];
    float*  zin  = (float*) (smem_raw);                              // 16 x 68 f32
    __bf16* xin  = (__bf16*)(smem_raw + SM_ZIN);                     // 16 x 72 bf16
    __bf16* hA   = (__bf16*)(smem_raw + SM_ZIN + SM_XIN);            // 16 x 520
    __bf16* hB   = (__bf16*)(smem_raw + SM_ZIN + SM_XIN + SM_H);     // 16 x 520
    float*  pbuf = (float*) (smem_raw + SM_ZIN + SM_XIN + 2*SM_H);   // 16 x 1480
    float*  ladsh= (float*) (smem_raw + SM_ZIN + SM_XIN + 2*SM_H + SM_P);

    const int tid  = threadIdx.x;
    const int row0 = blockIdx.x * 16;

    // ---- stage 0: load z tile + Afold (staged in pbuf area), LU transform ----
    float* zraw = pbuf;              // 16*68 f32
    float* Afl  = pbuf + 16 * 68;    // 4096 f32
    for (int o = tid; o < 16 * 64; o += 256) {
        int r = o >> 6, cdim = o & 63;
        zraw[r * 68 + cdim] = zsrc[(size_t)(row0 + r) * 64 + cdim];
    }
    for (int o = tid; o < 4096; o += 256) Afl[o] = Afold[o];
    if (tid < 16) ladsh[tid] = 0.0f;
    __syncthreads();

    for (int o = tid; o < 16 * 64; o += 256) {
        int r = o >> 6, c = o & 63;
        float s = lubias[c];
        for (int j = 0; j < 64; ++j)
            s += zraw[r * 68 + j] * Afl[j * 64 + c];
        zin[r * ZIN_STR + c] = s;
        xin[r * XIN_STR + c] = (__bf16)s;
    }
    __syncthreads();

    // ---- MADE network (all WMMA) ----
    made_gemm<true >(W0m, b0,  64,  32, xin, XIN_STR, hA, H_STR, nullptr, 0);
    __syncthreads();
    made_gemm<true >(Wh0, bh0, 512, 32, hA,  H_STR,   hB, H_STR, nullptr, 0);
    __syncthreads();
    made_gemm<true >(Wh1, bh1, 512, 32, hB,  H_STR,   hA, H_STR, nullptr, 0);
    __syncthreads();
    made_gemm<false>(Wom, bo,  512, 92, hA,  H_STR,   nullptr, 0, pbuf, P_STR);
    __syncthreads();

    // ---- rational-quadratic spline, one (row,dim) per task ----
    const float PADC = logf(expf(1.0f - MIND_) - 1.0f);
    for (int o = tid; o < 16 * 64; o += 256) {
        const int row = o >> 6, dim = o & 63;
        const float* pp = &pbuf[row * P_STR + dim * MULT_];

        float uw[8], uh[8], ud9[9];
#pragma unroll
        for (int j = 0; j < 8; ++j) uw[j] = pp[j];
#pragma unroll
        for (int j = 0; j < 8; ++j) uh[j] = pp[8 + j];
        ud9[0] = PADC; ud9[8] = PADC;
#pragma unroll
        for (int j = 0; j < 7; ++j) ud9[j + 1] = pp[16 + j];

        float cw[9], wv[8];
        {
            float mx = uw[0];
#pragma unroll
            for (int j = 1; j < 8; ++j) mx = fmaxf(mx, uw[j]);
            float e[8], s = 0.0f;
#pragma unroll
            for (int j = 0; j < 8; ++j) { e[j] = expf(uw[j] - mx); s += e[j]; }
            float inv = 1.0f / s, run = 0.0f;
            cw[0] = -TB_;
#pragma unroll
            for (int j = 0; j < 8; ++j) {
                run += MINW_ + (1.0f - MINW_ * 8.0f) * e[j] * inv;
                cw[j + 1] = 2.0f * TB_ * run - TB_;
            }
            cw[8] = TB_;
#pragma unroll
            for (int j = 0; j < 8; ++j) wv[j] = cw[j + 1] - cw[j];
        }
        float ch[9], hv[8];
        {
            float mx = uh[0];
#pragma unroll
            for (int j = 1; j < 8; ++j) mx = fmaxf(mx, uh[j]);
            float e[8], s = 0.0f;
#pragma unroll
            for (int j = 0; j < 8; ++j) { e[j] = expf(uh[j] - mx); s += e[j]; }
            float inv = 1.0f / s, run = 0.0f;
            ch[0] = -TB_;
#pragma unroll
            for (int j = 0; j < 8; ++j) {
                run += MINH_ + (1.0f - MINH_ * 8.0f) * e[j] * inv;
                ch[j + 1] = 2.0f * TB_ * run - TB_;
            }
            ch[8] = TB_;
#pragma unroll
            for (int j = 0; j < 8; ++j) hv[j] = ch[j + 1] - ch[j];
        }
        float dv[9];
#pragma unroll
        for (int j = 0; j < 9; ++j) dv[j] = MIND_ + softplusf_(ud9[j]);

        const float x = zin[row * ZIN_STR + dim];
        const bool inside = (x >= -TB_) && (x <= TB_);
        const float xc = fminf(fmaxf(x, -TB_), TB_);

        int idx = -1;
#pragma unroll
        for (int j = 0; j < 9; ++j) idx += (xc >= cw[j]) ? 1 : 0;
        idx = max(0, min(7, idx));

        float icw = 0, iw = 1, ich = 0, ih = 1, id0 = 1, id1 = 1;
#pragma unroll
        for (int j = 0; j < 8; ++j)
            if (j == idx) { icw = cw[j]; iw = wv[j]; ich = ch[j];
                            ih = hv[j]; id0 = dv[j]; id1 = dv[j + 1]; }

        const float idl = ih / iw;
        const float th  = (xc - icw) / iw;
        const float th1 = th * (1.0f - th);
        const float num = ih * (idl * th * th + id0 * th1);
        const float den = idl + (id0 + id1 - 2.0f * idl) * th1;
        const float outv = ich + num / den;
        const float omt = 1.0f - th;
        const float dnum = idl * idl * (id1 * th * th + 2.0f * idl * th1 + id0 * omt * omt);
        const float lad  = logf(dnum) - 2.0f * logf(den);

        zdst[(size_t)(row0 + row) * 64 + dim] = inside ? outv : x;
        atomicAdd(&ladsh[row], inside ? lad : 0.0f);
    }
    __syncthreads();

    if (tid < 16) {
        const int gr = row0 + tid;
        const float v = ladsh[tid];
        if (first) logq[gr] = v; else logq[gr] += v;
    }
}

// =====================================================================
// Finalize: logq += const terms - 0.5*D*log(2pi) - 0.5*||z||^2
// =====================================================================
__global__ void nsf_finalize(const float* __restrict__ zfin,
                             const float* __restrict__ logq,
                             const float* __restrict__ consts,
                             float* __restrict__ out)
{
    const int b = blockIdx.x * blockDim.x + threadIdx.x;
    float cs = 0.0f;
#pragma unroll
    for (int i = 0; i < NL_; ++i) cs += consts[i];
    float s = 0.0f;
    for (int d = 0; d < 64; ++d) {
        float v = zfin[(size_t)b * 64 + d];
        s += v * v;
    }
    out[b] = logq[b] + cs - 0.5f * 64.0f * logf(2.0f * 3.14159265358979f) - 0.5f * s;
}

// =====================================================================
extern "C" void kernel_launch(void* const* d_in, const int* in_sizes, int n_in,
                              void* d_out, int out_size, void* d_ws, size_t ws_size,
                              hipStream_t stream)
{
    const float* x    = (const float*)d_in[0];
    const float* W0   = (const float*)d_in[1];
    const float* b0   = (const float*)d_in[2];
    const float* Wh   = (const float*)d_in[3];
    const float* bh   = (const float*)d_in[4];
    const float* Wo   = (const float*)d_in[5];
    const float* bo   = (const float*)d_in[6];
    const float* lul  = (const float*)d_in[7];
    const float* luu  = (const float*)d_in[8];
    const float* lub  = (const float*)d_in[9];
    const int*   perm = (const int*)d_in[10];

    char* ws = (char*)d_ws;
    __bf16* W0m   = (__bf16*)(ws + OFF_W0M);
    __bf16* Whm   = (__bf16*)(ws + OFF_WHM);
    __bf16* Wom   = (__bf16*)(ws + OFF_WOM);
    float*  Afold = (float*) (ws + OFF_AFOLD);
    float*  consts= (float*) (ws + OFF_CONST);
    float*  logq  = (float*) (ws + OFF_LOGQ);
    float*  zA    = (float*) (ws + OFF_ZA);
    float*  zB    = (float*) (ws + OFF_ZB);

    (void)hipFuncSetAttribute((const void*)nsf_layer,
                              hipFuncAttributeMaxDynamicSharedMemorySize,
                              SMEM_TOTAL);

    nsf_prep_weights<<<2048, 256, 0, stream>>>(W0, Wh, Wo, W0m, Whm, Wom);
    nsf_prep_lu<<<NL_, 64, 0, stream>>>(lul, luu, perm, Afold, consts);

    const float* cur = x;
    float* nxt = zA;
    for (int i = NL_ - 1; i >= 0; --i) {
        nsf_layer<<<(int)(B_ / 16), 256, SMEM_TOTAL, stream>>>(
            cur, nxt,
            W0m + (size_t)i * H_ * D_,
            Whm + (size_t)(i * 2 + 0) * H_ * H_,
            Whm + (size_t)(i * 2 + 1) * H_ * H_,
            Wom + (size_t)i * P_ * H_,
            b0 + (size_t)i * H_,
            bh + (size_t)(i * 2 + 0) * H_,
            bh + (size_t)(i * 2 + 1) * H_,
            bo + (size_t)i * P_,
            Afold + (size_t)i * 4096,
            lub + (size_t)i * 64,
            logq, (i == NL_ - 1) ? 1 : 0);
        cur = nxt;
        nxt = (nxt == zA) ? zB : zA;
    }

    nsf_finalize<<<(int)(B_ / 256), 256, 0, stream>>>(cur, logq, consts, (float*)d_out);
}
